// CoAttention_Simple_3487513445084
// MI455X (gfx1250) — matrix operand order repaired and checked by reading, not compiled, not run on hardware
//
#include <hip/hip_runtime.h>

// ---------------------------------------------------------------------------
// CDNA5 / gfx1250 implementation of CoAttention_Simple.
// Heavy math: v_wmma_f32_16x16x32_f16 (wave32 WMMA).
// GEMM operand staging: Tensor Data Mover (tensor_load_to_lds) double-buffered
// with s_wait_tensorcnt, per CDNA5 ISA.
// ---------------------------------------------------------------------------

typedef __attribute__((ext_vector_type(16))) _Float16 v16h;
typedef __attribute__((ext_vector_type(8)))  _Float16 h8;
typedef __attribute__((ext_vector_type(4)))  _Float16 h4;
typedef __attribute__((ext_vector_type(8)))  float    v8f;
typedef __attribute__((ext_vector_type(4)))  unsigned int u32x4;
typedef __attribute__((ext_vector_type(8)))  int      i32x8;
typedef __attribute__((ext_vector_type(4)))  int      i32x4;

#if defined(__has_builtin)
#if __has_builtin(__builtin_amdgcn_tensor_load_to_lds) && \
    __has_builtin(__builtin_amdgcn_s_wait_tensorcnt)
#define USE_TDM 1
#endif
#endif
#ifndef USE_TDM
#define USE_TDM 0
#endif

__device__ inline v8f vzero8() {
  v8f v;
#pragma unroll
  for (int i = 0; i < 8; ++i) v[i] = 0.f;
  return v;
}

__device__ inline v16h frag_cat(h8 lo, h8 hi) {
  return __builtin_shufflevector(lo, hi, 0,1,2,3,4,5,6,7,8,9,10,11,12,13,14,15);
}

// A-matrix fragment (16x32 f16, MxK): lane&15 = row M; lanes 0-15 hold
// K 0-15, lanes 16-31 hold K 16-31 (matches B fragment).
__device__ inline v16h load_a_frag(const _Float16* row, int lane) {
  const int lk = (lane & 16) ? 16 : 0;
  h8 lo = *(const h8*)(row + lk);
  h8 hi = *(const h8*)(row + lk + 8);
  return frag_cat(lo, hi);
}

// B-matrix fragment (32x16 f16, KxN): lane&15 = column N; lanes 0-15 hold
// K 0-15, lanes 16-31 hold K 16-31.
__device__ inline v16h load_b_frag(const _Float16* p) {
  h8 lo = *(const h8*)(p);
  h8 hi = *(const h8*)(p + 8);
  return frag_cat(lo, hi);
}

__device__ inline v8f wmma_f16(v16h a, v16h b, v8f c) {
  return __builtin_amdgcn_wmma_f32_16x16x32_f16(false, a, false, b, (short)0, c,
                                                false, false);
}

// Stage 16 source elements (fp32 or f16) into LDS as f16, vectorized.
__device__ inline void stage16(const float* __restrict__ s, _Float16* d) {
  const float4* sv = (const float4*)s;
  float4 f0 = sv[0], f1 = sv[1], f2 = sv[2], f3 = sv[3];
  h8 lo, hi;
  lo[0] = (_Float16)f0.x; lo[1] = (_Float16)f0.y; lo[2] = (_Float16)f0.z; lo[3] = (_Float16)f0.w;
  lo[4] = (_Float16)f1.x; lo[5] = (_Float16)f1.y; lo[6] = (_Float16)f1.z; lo[7] = (_Float16)f1.w;
  hi[0] = (_Float16)f2.x; hi[1] = (_Float16)f2.y; hi[2] = (_Float16)f2.z; hi[3] = (_Float16)f2.w;
  hi[4] = (_Float16)f3.x; hi[5] = (_Float16)f3.y; hi[6] = (_Float16)f3.z; hi[7] = (_Float16)f3.w;
  *(h8*)d = lo;
  *((h8*)d + 1) = hi;
}
__device__ inline void stage16(const _Float16* __restrict__ s, _Float16* d) {
  const h8* sv = (const h8*)s;
  *(h8*)d = sv[0];
  *((h8*)d + 1) = sv[1];
}

#define TILE_M 128
#define TILE_N 128
#define TILE_K 32
#define LDSS   40   // 32 data + 8 pad f16; 40 % 8 == 0 keeps 16B alignment

// ---------------------------------------------------------------------------
// TDM tile descriptor: DMA a (rows=128) x (K=32) f16 tile, row stride = K
// elements, into LDS with 4-DWORD padding after every 16 DWORDs (one row),
// reproducing the [128][LDSS] layout.  ISA 08_async_tensor §8.3/§8.4.
// ---------------------------------------------------------------------------
#if USE_TDM
__device__ inline void tdm_issue_tile(const _Float16* base, int row0, int nrows,
                                      int K, int k0, unsigned ldsOff) {
  unsigned long long ga =
      (unsigned long long)(size_t)base + ((unsigned long long)row0 * K + k0) * 2;
  u32x4 g0;
  g0[0] = 1u;                                   // count=1, user descriptor
  g0[1] = ldsOff;                               // lds_addr (bytes)
  g0[2] = (unsigned)ga;                         // global_addr[31:0]
  g0[3] = (unsigned)((ga >> 32) & 0x01FFFFFFu)  // global_addr[56:32]
          | (2u << 30);                         // type = 2 ("image")
  i32x8 g1;
  // mask=0 | data_size=1(2B)<<16 | pad_enable<<20 | pad_interval=3(16dw)<<22
  // | pad_amount=3(4dw)<<25
  g1[0] = (int)((1u << 16) | (1u << 20) | (3u << 22) | (3u << 25));
  g1[1] = (int)(((unsigned)K & 0xFFFFu) << 16);            // tensor_dim0[15:0]
  g1[2] = (int)(((unsigned)K >> 16) |                      // tensor_dim0[31:16]
                (((unsigned)nrows & 0xFFFFu) << 16));      // tensor_dim1[15:0]
  g1[3] = (int)(((unsigned)nrows >> 16) | (32u << 16));    // dim1 hi | tile_dim0=32
  g1[4] = (int)128u;                                       // tile_dim1=128, tile_dim2=0
  g1[5] = (int)(unsigned)K;                                // tensor_dim0_stride lo32
  g1[6] = 0;                                               // stride0 hi | stride1 lo
  g1[7] = 0;
  i32x4 z4;
  z4[0] = 0; z4[1] = 0; z4[2] = 0; z4[3] = 0;
  i32x8 z8;
#pragma unroll
  for (int i = 0; i < 8; ++i) z8[i] = 0;
  __builtin_amdgcn_tensor_load_to_lds(g0, g1, z4, z4, z8, 0);
}
#endif

// ---------------------------------------------------------------------------
// TDM GEMM: C(MxN) = (A(MxK) @ W(NxK)^T + bias) * scale, A and W f16.
// Double-buffered LDS tiles fed by the Tensor Data Mover; wave 0 drives the
// DMA and TENSORcnt, barriers publish completion to the workgroup.
// EPI: 0 -> f16 store; 2 -> fp32 store.
// ---------------------------------------------------------------------------
template <int EPI>
__global__ void __launch_bounds__(256)
gemm_tdm_kernel(const _Float16* __restrict__ Ap, const _Float16* __restrict__ Wp,
                const float* __restrict__ bias, void* __restrict__ Cout,
                int M, int N, int K, float scale) {
  __shared__ _Float16 As[2][TILE_M][LDSS];
  __shared__ _Float16 Ws[2][TILE_N][LDSS];

  const int tid  = threadIdx.x;
  const int lane = tid & 31;
  const int wid  = tid >> 5;
  const int lcol = lane & 15;
  const int bm = blockIdx.y * TILE_M;
  const int bn = blockIdx.x * TILE_N;
  const int wm = (wid >> 2) * 64;
  const int wn = (wid & 3) * 32;

  v8f acc[4][2];
#pragma unroll
  for (int mi = 0; mi < 4; ++mi)
#pragma unroll
    for (int ni = 0; ni < 2; ++ni) acc[mi][ni] = vzero8();

#if USE_TDM
  const unsigned ldsA[2] = {(unsigned)(size_t)&As[0][0][0],
                            (unsigned)(size_t)&As[1][0][0]};
  const unsigned ldsW[2] = {(unsigned)(size_t)&Ws[0][0][0],
                            (unsigned)(size_t)&Ws[1][0][0]};
  if (wid == 0) {  // prologue: DMA first tiles into buffer 0
    tdm_issue_tile(Ap, bm, M, K, 0, ldsA[0]);
    tdm_issue_tile(Wp, bn, N, K, 0, ldsW[0]);
  }
#else
  const int r    = tid >> 1;
  const int half = (tid & 1) * 16;
  const _Float16* aPtr = Ap + (size_t)(bm + r) * K + half;
  const _Float16* wPtr = Wp + (size_t)(bn + r) * K + half;
#endif

  int buf = 0;
  for (int k0 = 0; k0 < K; k0 += TILE_K) {
#if USE_TDM
    const bool more = (k0 + TILE_K) < K;
    if (wid == 0) {
      if (more) {  // overlap: DMA next tiles into the other buffer
        tdm_issue_tile(Ap, bm, M, K, k0 + TILE_K, ldsA[buf ^ 1]);
        tdm_issue_tile(Wp, bn, N, K, k0 + TILE_K, ldsW[buf ^ 1]);
        __builtin_amdgcn_s_wait_tensorcnt(2);  // release current tiles
      } else {
        __builtin_amdgcn_s_wait_tensorcnt(0);
      }
    }
    __syncthreads();  // publish DMA completion to all waves
#else
    __syncthreads();
    stage16(aPtr + k0, &As[buf][tid >> 1][(tid & 1) * 16]);
    stage16(wPtr + k0, &Ws[buf][tid >> 1][(tid & 1) * 16]);
    __syncthreads();
#endif

    v16h af[4], bf[2];
#pragma unroll
    for (int mi = 0; mi < 4; ++mi)
      af[mi] = load_a_frag(&As[buf][wm + mi * 16 + lcol][0], lane);
#pragma unroll
    for (int ni = 0; ni < 2; ++ni)
      bf[ni] = load_b_frag(&Ws[buf][wn + ni * 16 + lcol][(lane & 16) ? 16 : 0]);

#pragma unroll
    for (int mi = 0; mi < 4; ++mi)
#pragma unroll
      for (int ni = 0; ni < 2; ++ni)
        acc[mi][ni] = wmma_f16(af[mi], bf[ni], acc[mi][ni]);

    __syncthreads();  // all reads of this buffer done before DMA reuses it
    buf ^= 1;
  }

  const int rowAdd = (lane & 16) ? 8 : 0;
#pragma unroll
  for (int ni = 0; ni < 2; ++ni) {
    const int gn = bn + wn + ni * 16 + lcol;
    const float bv = bias ? bias[gn] : 0.f;
#pragma unroll
    for (int mi = 0; mi < 4; ++mi) {
#pragma unroll
      for (int j = 0; j < 8; ++j) {
        const int gm = bm + wm + mi * 16 + rowAdd + j;
        const float v = (acc[mi][ni][j] + bv) * scale;
        if (EPI == 0) {
          ((_Float16*)Cout)[(size_t)gm * N + gn] = (_Float16)v;
        } else {
          ((float*)Cout)[(size_t)gm * N + gn] = v;
        }
      }
    }
  }
}

// ---------------------------------------------------------------------------
// Synchronous-staging GEMM for fp32 A / fp32 W (in_proj only).
// EPI 1: f16 store with in_proj permute (B,C,T,E)->(T,B,C*E).
// ---------------------------------------------------------------------------
template <typename AT, int EPI>
__global__ void __launch_bounds__(256)
gemm_kernel(const AT* __restrict__ Ap, const float* __restrict__ Wp,
            const float* __restrict__ bias, void* __restrict__ Cout,
            int M, int N, int K, float scale) {
  __shared__ _Float16 As[TILE_M][LDSS];
  __shared__ _Float16 Ws[TILE_N][LDSS];

  const int tid  = threadIdx.x;
  const int lane = tid & 31;
  const int wid  = tid >> 5;
  const int lcol = lane & 15;
  const int bm = blockIdx.y * TILE_M;
  const int bn = blockIdx.x * TILE_N;
  const int wm = (wid >> 2) * 64;
  const int wn = (wid & 3) * 32;

  v8f acc[4][2];
#pragma unroll
  for (int mi = 0; mi < 4; ++mi)
#pragma unroll
    for (int ni = 0; ni < 2; ++ni) acc[mi][ni] = vzero8();

  const int r    = tid >> 1;
  const int half = (tid & 1) * 16;
  const AT*    aPtr = Ap + (size_t)(bm + r) * K + half;
  const float* wPtr = Wp + (size_t)(bn + r) * K + half;

  for (int k0 = 0; k0 < K; k0 += TILE_K) {
    __syncthreads();
    stage16(aPtr + k0, &As[r][half]);
    stage16(wPtr + k0, &Ws[r][half]);
    __syncthreads();

    v16h af[4], bf[2];
#pragma unroll
    for (int mi = 0; mi < 4; ++mi)
      af[mi] = load_a_frag(&As[wm + mi * 16 + lcol][0], lane);
#pragma unroll
    for (int ni = 0; ni < 2; ++ni)
      bf[ni] = load_b_frag(&Ws[wn + ni * 16 + lcol][(lane & 16) ? 16 : 0]);

#pragma unroll
    for (int mi = 0; mi < 4; ++mi)
#pragma unroll
      for (int ni = 0; ni < 2; ++ni)
        acc[mi][ni] = wmma_f16(af[mi], bf[ni], acc[mi][ni]);
  }

  const int rowAdd = (lane & 16) ? 8 : 0;
#pragma unroll
  for (int ni = 0; ni < 2; ++ni) {
    const int gn = bn + wn + ni * 16 + lcol;
    const float bv = bias ? bias[gn] : 0.f;
#pragma unroll
    for (int mi = 0; mi < 4; ++mi) {
#pragma unroll
      for (int j = 0; j < 8; ++j) {
        const int gm = bm + wm + mi * 16 + rowAdd + j;
        const float v = (acc[mi][ni][j] + bv) * scale;
        if (EPI == 0) {
          ((_Float16*)Cout)[(size_t)gm * N + gn] = (_Float16)v;
        } else if (EPI == 1) {
          const int t = gm & 1023;
          const int c = (gm >> 10) & 3;
          const int b = gm >> 12;
          ((_Float16*)Cout)[((size_t)(t * 8 + b)) * 2048 + c * 512 + gn] =
              (_Float16)v;
        } else {
          ((float*)Cout)[(size_t)gm * N + gn] = v;
        }
      }
    }
  }
}

// ---------------------------------------------------------------------------
// fp32 -> f16 weight conversion (4 elements/thread, vectorized)
// ---------------------------------------------------------------------------
__global__ void __launch_bounds__(256)
cvt_f16_kernel(const float* __restrict__ in, _Float16* __restrict__ out, long n) {
  long i = ((long)blockIdx.x * 256 + threadIdx.x) * 4;
  if (i >= n) return;
  float4 f = *(const float4*)(in + i);
  h4 o;
  o[0] = (_Float16)f.x; o[1] = (_Float16)f.y;
  o[2] = (_Float16)f.z; o[3] = (_Float16)f.w;
  *(h4*)(out + i) = o;
}

// ---------------------------------------------------------------------------
// GLU: in (rows x 2H) f16 -> out (rows x H) f16; out = a * sigmoid(b)
// ---------------------------------------------------------------------------
__global__ void __launch_bounds__(256)
glu_kernel(const _Float16* __restrict__ in, _Float16* __restrict__ out,
           long rows, long H) {
  long idx = (long)blockIdx.x * blockDim.x + threadIdx.x;
  long n = rows * H;
  if (idx >= n) return;
  long rr = idx / H, j = idx % H;
  float a = (float)in[(size_t)rr * 2 * H + j];
  float g = (float)in[(size_t)rr * 2 * H + H + j];
  out[idx] = (_Float16)(a / (1.f + __expf(-g)));
}

// ---------------------------------------------------------------------------
// Flash attention.  q/k/v: f16 (T*B, A), A = H*HD.  Grid (B*H, T/128);
// 8 waves, each wave owns 16 query rows.  Softmax scale folded into q.
// ---------------------------------------------------------------------------
__global__ void __launch_bounds__(256)
attn_kernel(const _Float16* __restrict__ qb, const _Float16* __restrict__ kb,
            const _Float16* __restrict__ vb, _Float16* __restrict__ ob) {
  constexpr int T = 1024, NB = 8, NH = 8, HD = 256, AD = 2048;
  __shared__ _Float16 Ks[32][HD + 8];
  __shared__ _Float16 Vt[HD][LDSS];
  __shared__ _Float16 Ps[8][16][LDSS];

  const int tid  = threadIdx.x;
  const int lane = tid & 31;
  const int wid  = tid >> 5;
  const int lcol = lane & 15;
  const int kh   = (lane & 16) ? 16 : 0;
  const int hb = blockIdx.x;
  const int b = hb / NH, h = hb % NH;
  const int t0 = blockIdx.y * 128 + wid * 16;

  v16h qf[8];
  {
    const _Float16* qrow = qb + ((size_t)(t0 + lcol) * NB + b) * AD + h * HD;
#pragma unroll
    for (int kc = 0; kc < 8; ++kc) qf[kc] = load_a_frag(qrow + kc * 32, lane);
  }

  v8f oacc[16];
#pragma unroll
  for (int tt = 0; tt < 16; ++tt) oacc[tt] = vzero8();
  float rm[8], rl[8];
#pragma unroll
  for (int j = 0; j < 8; ++j) { rm[j] = -3.0e38f; rl[j] = 0.f; }

  const int r8 = tid >> 3;
  const int s8 = (tid & 7) * 32;

  for (int tk0 = 0; tk0 < T; tk0 += 32) {
    __syncthreads();
    {
      const _Float16* src = kb + ((size_t)(tk0 + r8) * NB + b) * AD + h * HD + s8;
      const h8* sv = (const h8*)src;
      h8* dv = (h8*)&Ks[r8][s8];
      dv[0] = sv[0]; dv[1] = sv[1]; dv[2] = sv[2]; dv[3] = sv[3];
    }
    {
      const _Float16* src = vb + ((size_t)(tk0 + r8) * NB + b) * AD + h * HD + s8;
      h8 v0 = ((const h8*)src)[0], v1 = ((const h8*)src)[1];
      h8 v2 = ((const h8*)src)[2], v3 = ((const h8*)src)[3];
#pragma unroll
      for (int i = 0; i < 8; ++i) Vt[s8 + i][r8]      = v0[i];
#pragma unroll
      for (int i = 0; i < 8; ++i) Vt[s8 + 8 + i][r8]  = v1[i];
#pragma unroll
      for (int i = 0; i < 8; ++i) Vt[s8 + 16 + i][r8] = v2[i];
#pragma unroll
      for (int i = 0; i < 8; ++i) Vt[s8 + 24 + i][r8] = v3[i];
    }
    __syncthreads();

    v8f s0 = vzero8(), s1 = vzero8();
#pragma unroll
    for (int kc = 0; kc < 8; ++kc) {
      v16h b0 = load_b_frag(&Ks[lcol][kc * 32 + kh]);
      v16h b1 = load_b_frag(&Ks[16 + lcol][kc * 32 + kh]);
      s0 = wmma_f16(qf[kc], b0, s0);
      s1 = wmma_f16(qf[kc], b1, s1);
    }

#pragma unroll
    for (int j = 0; j < 8; ++j) {
      float mx = fmaxf(s0[j], s1[j]);
      mx = fmaxf(mx, __shfl_xor(mx, 1));
      mx = fmaxf(mx, __shfl_xor(mx, 2));
      mx = fmaxf(mx, __shfl_xor(mx, 4));
      mx = fmaxf(mx, __shfl_xor(mx, 8));
      const float nm = fmaxf(rm[j], mx);
      const float al = __expf(rm[j] - nm);
      const float p0 = __expf(s0[j] - nm);
      const float p1 = __expf(s1[j] - nm);
      float rs = p0 + p1;
      rs += __shfl_xor(rs, 1);
      rs += __shfl_xor(rs, 2);
      rs += __shfl_xor(rs, 4);
      rs += __shfl_xor(rs, 8);
      rl[j] = rl[j] * al + rs;
      rm[j] = nm;
#pragma unroll
      for (int tt = 0; tt < 16; ++tt) oacc[tt][j] *= al;
      const int pr = j + ((lane & 16) ? 8 : 0);
      Ps[wid][pr][lcol]      = (_Float16)p0;
      Ps[wid][pr][16 + lcol] = (_Float16)p1;
    }

    v16h pf = load_a_frag(&Ps[wid][lcol][0], lane);

#pragma unroll
    for (int tt = 0; tt < 16; ++tt) {
      v16h vf = load_b_frag(&Vt[tt * 16 + lcol][kh]);
      oacc[tt] = wmma_f16(pf, vf, oacc[tt]);
    }
  }

#pragma unroll
  for (int tt = 0; tt < 16; ++tt) {
#pragma unroll
    for (int j = 0; j < 8; ++j) {
      const int t = t0 + j + ((lane & 16) ? 8 : 0);
      const int d = tt * 16 + lcol;
      const float v = oacc[tt][j] / rl[j];
      ob[((size_t)t * NB + b) * AD + h * HD + d] = (_Float16)v;
    }
  }
}

// ---------------------------------------------------------------------------
// Finalize: z_c = attnproj + 2*x[:,c]; LayerNorm(D) per c; mean over C.
// ---------------------------------------------------------------------------
__device__ inline float block_sum(float v, float* red, int lane, int wid) {
#pragma unroll
  for (int m = 16; m > 0; m >>= 1) v += __shfl_xor(v, m);
  __syncthreads();
  if (lane == 0) red[wid] = v;
  __syncthreads();
  float t = 0.f;
#pragma unroll
  for (int i = 0; i < 8; ++i) t += red[i];
  return t;
}

__global__ void __launch_bounds__(256)
finalize_kernel(const float* __restrict__ ap, const float* __restrict__ xp,
                const float* __restrict__ lng, const float* __restrict__ lnb,
                float* __restrict__ outp) {
  constexpr int T = 1024, D = 512, NB = 8, C = 4;
  __shared__ float red[8];
  const int bt = blockIdx.x;
  const int b = bt / T, t = bt % T;
  const int tid = threadIdx.x, lane = tid & 31, wid = tid >> 5;

  const float a0 = ap[((size_t)t * NB + b) * D + tid];
  const float a1 = ap[((size_t)t * NB + b) * D + tid + 256];
  const float g0 = lng[tid], g1 = lng[tid + 256];
  const float e0 = lnb[tid], e1 = lnb[tid + 256];

  float acc0 = 0.f, acc1 = 0.f;
  for (int c = 0; c < C; ++c) {
    const float* xr = xp + ((size_t)(b * C + c) * T + t) * D;
    const float z0 = a0 + 2.f * xr[tid];
    const float z1 = a1 + 2.f * xr[tid + 256];
    const float mu = block_sum(z0 + z1, red, lane, wid) * (1.f / 512.f);
    const float d0 = z0 - mu, d1 = z1 - mu;
    const float var = block_sum(d0 * d0 + d1 * d1, red, lane, wid) * (1.f / 512.f);
    const float rstd = rsqrtf(var + 1e-5f);
    acc0 += d0 * rstd * g0 + e0;
    acc1 += d1 * rstd * g1 + e1;
  }
  outp[(size_t)bt * D + tid]       = acc0 * 0.25f;
  outp[(size_t)bt * D + tid + 256] = acc1 * 0.25f;
}

// ---------------------------------------------------------------------------
// Orchestration
// ---------------------------------------------------------------------------
extern "C" void kernel_launch(void* const* d_in, const int* in_sizes, int n_in,
                              void* d_out, int out_size, void* d_ws,
                              size_t ws_size, hipStream_t stream) {
  (void)in_sizes; (void)n_in; (void)out_size; (void)ws_size;

  const float* x  = (const float*)d_in[0];
  const float* Wq = (const float*)d_in[1];
  const float* bq = (const float*)d_in[2];
  const float* Wk = (const float*)d_in[3];
  const float* bk = (const float*)d_in[4];
  const float* Wv = (const float*)d_in[5];
  const float* bv = (const float*)d_in[6];
  const float* gW[3][3], *gB[3][3];
  for (int s = 0; s < 3; ++s)
    for (int l = 0; l < 3; ++l) {
      gW[s][l] = (const float*)d_in[7 + s * 6 + l * 2];
      gB[s][l] = (const float*)d_in[7 + s * 6 + l * 2 + 1];
    }
  const float* Wo  = (const float*)d_in[25];
  const float* bo  = (const float*)d_in[26];
  const float* lng = (const float*)d_in[27];
  const float* lnb = (const float*)d_in[28];

  // workspace layout (~418 MB)
  constexpr size_t MB = 1ull << 20;
  char* ws = (char*)d_ws;
  _Float16* qkv[3] = {(_Float16*)(ws + 0 * MB), (_Float16*)(ws + 32 * MB),
                      (_Float16*)(ws + 64 * MB)};        // 32 MB each
  _Float16* h1   = (_Float16*)(ws + 96 * MB);            // 128 MB
  _Float16* h1g  = (_Float16*)(ws + 224 * MB);           // 64 MB
  _Float16* h2   = (_Float16*)(ws + 288 * MB);           // 64 MB
  _Float16* h2g  = (_Float16*)(ws + 352 * MB);           // 32 MB
  _Float16* wbuf = (_Float16*)(ws + 384 * MB);           // 34 MB (f16 weights)
  _Float16* attnout = h1;                                // reuse
  float* attnproj = (float*)(ws + 224 * MB);             // reuse h1g

  const dim3 blk(256);
  const float HD_SCALE = 0.0625f;  // 256^-0.5

  // --- in_proj + permute: fp32 path ---
  const float* Wproj[3] = {Wq, Wk, Wv};
  const float* bproj[3] = {bq, bk, bv};
  for (int s = 0; s < 3; ++s)
    gemm_kernel<float, 1><<<dim3(512 / TILE_N, 32768 / TILE_M), blk, 0, stream>>>(
        x, Wproj[s], bproj[s], qkv[s], 32768, 512, 512, 1.f);

  // --- gated linear stacks (TDM GEMMs; f16 weights staged in wbuf) ---
  for (int s = 0; s < 3; ++s) {
    const float sc = (s == 0) ? HD_SCALE : 1.f;
    {
      long n = 8192L * 2048L;
      cvt_f16_kernel<<<dim3((unsigned)((n / 4 + 255) / 256)), blk, 0, stream>>>(gW[s][0], wbuf, n);
      gemm_tdm_kernel<0><<<dim3(8192 / TILE_N, 8192 / TILE_M), blk, 0, stream>>>(
          qkv[s], wbuf, gB[s][0], h1, 8192, 8192, 2048, 1.f);
    }
    {
      long n = 8192L * 4096L;
      glu_kernel<<<dim3((unsigned)((n + 255) / 256)), blk, 0, stream>>>(h1, h1g, 8192, 4096);
    }
    {
      long n = 4096L * 4096L;
      cvt_f16_kernel<<<dim3((unsigned)((n / 4 + 255) / 256)), blk, 0, stream>>>(gW[s][1], wbuf, n);
      gemm_tdm_kernel<0><<<dim3(4096 / TILE_N, 8192 / TILE_M), blk, 0, stream>>>(
          h1g, wbuf, gB[s][1], h2, 8192, 4096, 4096, 1.f);
    }
    {
      long n = 8192L * 2048L;
      glu_kernel<<<dim3((unsigned)((n + 255) / 256)), blk, 0, stream>>>(h2, h2g, 8192, 2048);
    }
    {
      long n = 2048L * 2048L;
      cvt_f16_kernel<<<dim3((unsigned)((n / 4 + 255) / 256)), blk, 0, stream>>>(gW[s][2], wbuf, n);
      gemm_tdm_kernel<0><<<dim3(2048 / TILE_N, 8192 / TILE_M), blk, 0, stream>>>(
          h2g, wbuf, gB[s][2], qkv[s], 8192, 2048, 2048, sc);
    }
  }

  // --- attention ---
  attn_kernel<<<dim3(64, 8), blk, 0, stream>>>(qkv[0], qkv[1], qkv[2], attnout);

  // --- output projection (TDM GEMM, fp32 out) ---
  {
    long n = 512L * 2048L;
    cvt_f16_kernel<<<dim3((unsigned)((n / 4 + 255) / 256)), blk, 0, stream>>>(Wo, wbuf, n);
    gemm_tdm_kernel<2><<<dim3(512 / TILE_N, 8192 / TILE_M), blk, 0, stream>>>(
        attnout, wbuf, bo, attnproj, 8192, 512, 2048, 1.f);
  }

  // --- residual + LN + mean over C ---
  finalize_kernel<<<dim3(8 * 1024), blk, 0, stream>>>(attnproj, x, lng, lnb,
                                                      (float*)d_out);
}
